// LinearAttend_66125316489931
// MI455X (gfx1250) — matrix operand order repaired and checked by reading, not compiled, not run on hardware
//
#include <hip/hip_runtime.h>
#include <math.h>

typedef __attribute__((ext_vector_type(16))) _Float16 v16h;
typedef __attribute__((ext_vector_type(8)))  _Float16 v8h;
typedef __attribute__((ext_vector_type(4)))  _Float16 v4h;
typedef __attribute__((ext_vector_type(8)))  float    v8f;

#define NHEADS 32      // B*N = 4*8
#define DHEAD  64
#define SEQ    8192
#define KC     128     // K-chunk staged in LDS per step (context kernel)
#define SCHUNK 512     // seq columns per workgroup (context kernel)
#define LDSTR  (KC + 8)  // 136 halves = 272 B row stride (16B- and 8B-aligned)
#define SC     256     // seq columns per workgroup (output kernel)

// ---------------- kernel 0: zero fp32 context accumulator ----------------
__global__ void zero_ctx_kernel(float* __restrict__ ctx, int n) {
    int i = blockIdx.x * blockDim.x + threadIdx.x;
    if (i < n) ctx[i] = 0.f;
}

// ------- kernel 1: per-row stats of k: rowmax and 1/sum(exp(k-max)) -------
__global__ __launch_bounds__(256) void kstats_kernel(const float* __restrict__ k,
                                                     float* __restrict__ kmax,
                                                     float* __restrict__ kinv) {
    const int row = blockIdx.x;                      // [0, 2048) = head*64 + d
    const float4* kr = (const float4*)(k + (size_t)row * SEQ);
    float m = -INFINITY, s = 0.f;
    for (int i = threadIdx.x; i < SEQ / 4; i += 256) {   // 8 x float4 per thread
        float4 x4 = kr[i];
        #pragma unroll
        for (int e = 0; e < 4; ++e) {
            float x  = (&x4.x)[e];
            float nm = fmaxf(m, x);
            s = s * __expf(m - nm) + __expf(x - nm);
            m = nm;
        }
    }
    __shared__ float sm[256], ss[256];
    sm[threadIdx.x] = m; ss[threadIdx.x] = s;
    __syncthreads();
    for (int off = 128; off > 0; off >>= 1) {
        if (threadIdx.x < off) {
            float m2 = sm[threadIdx.x + off], s2 = ss[threadIdx.x + off];
            float nm = fmaxf(sm[threadIdx.x], m2);
            ss[threadIdx.x] = ss[threadIdx.x] * __expf(sm[threadIdx.x] - nm) + s2 * __expf(m2 - nm);
            sm[threadIdx.x] = nm;
        }
        __syncthreads();
    }
    if (threadIdx.x == 0) { kmax[row] = sm[0]; kinv[row] = 1.f / ss[0]; }
}

// ------- kernel 2: context[h,i,j] += sum_s ks[i,s] * v[j,s]  (WMMA f16) -------
__global__ __launch_bounds__(256) void context_kernel(const float* __restrict__ k,
                                                      const float* __restrict__ v,
                                                      const float* __restrict__ kmax,
                                                      const float* __restrict__ kinv,
                                                      float* __restrict__ ctx) {
    const int head   = blockIdx.x;                   // 32
    const int schunk = blockIdx.y;                   // SEQ/SCHUNK = 16
    const size_t base = (size_t)head * DHEAD * SEQ;
    const int s0base  = schunk * SCHUNK;

    __shared__ _Float16 Alds[DHEAD][LDSTR];          // ks  : row i, cols s_local
    __shared__ _Float16 Blds[DHEAD][LDSTR];          // v   : row j, cols s_local

    const int tid  = threadIdx.x;
    const int lane = tid & 31;
    const int wave = tid >> 5;
    const int half = lane >> 4;
    const int l16  = lane & 15;
    const int ti   = wave >> 1;                      // i-tile [0,4)
    const int tj0  = (2 * wave) & 3;                 // j-tiles (even, even+1)
    const int tj1  = tj0 + 1;

    v8f acc0 = {}; v8f acc1 = {};

    for (int s0 = s0base; s0 < s0base + SCHUNK; s0 += KC) {
        __syncthreads();
        // vectorized stage: float4 global loads, packed 4xf16 LDS stores.
        // 64 rows x 32 float4 = 2048 vectors, 8 per thread.
        for (int idx = tid; idx < DHEAD * (KC / 4); idx += 256) {
            int r   = idx >> 5;                      // KC/4 == 32 vectors per row
            int c   = (idx & 31) * 4;
            int grow = head * DHEAD + r;
            float mx = kmax[grow], iv = kinv[grow];
            float4 kx = *(const float4*)&k[base + (size_t)r * SEQ + s0 + c];
            float4 vx = *(const float4*)&v[base + (size_t)r * SEQ + s0 + c];
            v4h ka, va;
            #pragma unroll
            for (int e = 0; e < 4; ++e) {
                ka[e] = (_Float16)(__expf((&kx.x)[e] - mx) * iv);
                va[e] = (_Float16)((&vx.x)[e]);
            }
            *(v4h*)&Alds[r][c] = ka;                 // 8B store, 8B aligned
            *(v4h*)&Blds[r][c] = va;
        }
        __syncthreads();

        #pragma unroll
        for (int kk = 0; kk < KC; kk += 32) {
            // A (16x32 f16): lane<16 -> M=l16, K {kk..kk+7, kk+16..kk+23}; lane>=16 shifted by 8
            const _Float16* ap = &Alds[16 * ti + l16][kk + 8 * half];
            v8h alo = *(const v8h*)ap;
            v8h ahi = *(const v8h*)(ap + 16);
            v16h A;
            #pragma unroll
            for (int e = 0; e < 8; ++e) { A[e] = alo[e]; A[e + 8] = ahi[e]; }

            // B (32x16 f16): lane<16 -> N=l16, K {kk..kk+7, kk+16..kk+23}; lane>=16 shifted by 8
            const _Float16* bp0 = &Blds[16 * tj0 + l16][kk + 8 * half];
            const _Float16* bp1 = &Blds[16 * tj1 + l16][kk + 8 * half];
            v8h b0lo = *(const v8h*)bp0, b0hi = *(const v8h*)(bp0 + 16);
            v8h b1lo = *(const v8h*)bp1, b1hi = *(const v8h*)(bp1 + 16);
            v16h B0, B1;
            #pragma unroll
            for (int e = 0; e < 8; ++e) {
                B0[e] = b0lo[e]; B0[e + 8] = b0hi[e];
                B1[e] = b1lo[e]; B1[e + 8] = b1hi[e];
            }

            acc0 = __builtin_amdgcn_wmma_f32_16x16x32_f16(false, A, false, B0, (short)0, acc0, false, false);
            acc1 = __builtin_amdgcn_wmma_f32_16x16x32_f16(false, A, false, B1, (short)0, acc1, false, false);
        }
    }

    float* ctxh = ctx + head * (DHEAD * DHEAD);
    #pragma unroll
    for (int r = 0; r < 8; ++r) {
        int i = 16 * ti + 8 * half + r;              // C/D layout: VGPR r -> M = r + 8*half
        atomicAdd(&ctxh[i * DHEAD + 16 * tj0 + l16], acc0[r]);
        atomicAdd(&ctxh[i * DHEAD + 16 * tj1 + l16], acc1[r]);
    }
}

// ------- kernel 3: out[h,j,s] = sum_i ctx[i,j] * qs[i,s]  (WMMA f16) -------
__global__ __launch_bounds__(256) void output_kernel(const float* __restrict__ q,
                                                     const float* __restrict__ ctx,
                                                     float* __restrict__ out) {
    const int head = blockIdx.x;                     // 32
    const int sch  = blockIdx.y;                     // SEQ/SC = 32
    const size_t base = (size_t)head * DHEAD * SEQ;
    const int s0 = sch * SC;

    __shared__ _Float16 Ac[DHEAD][72];               // ctx^T: [j][i], 9.2 KB
    __shared__ _Float16 Bq[SC][72];                  // qs   : [s_local][i], 36.9 KB

    const int tid = threadIdx.x;

    // ctx^T into LDS (fp16)
    for (int idx = tid; idx < DHEAD * DHEAD; idx += 256) {
        int i = idx >> 6, j = idx & 63;
        Ac[j][i] = (_Float16)ctx[head * DHEAD * DHEAD + idx];
    }

    // column softmax over feature dim i: single HBM pass, column held in registers.
    {
        const float* qc = q + base + s0 + tid;       // one column per thread
        float qv[DHEAD];
        float m = -INFINITY, sum = 0.f;
        #pragma unroll
        for (int i = 0; i < DHEAD; ++i) {
            float x  = qc[(size_t)i * SEQ];
            qv[i] = x;
            float nm = fmaxf(m, x);
            sum = sum * __expf(m - nm) + __expf(x - nm);
            m = nm;
        }
        float sc = 0.125f / sum;                     // scale = D^-0.5 = 0.125
        #pragma unroll
        for (int i = 0; i < DHEAD; i += 4) {
            v4h h;
            #pragma unroll
            for (int e = 0; e < 4; ++e) h[e] = (_Float16)(__expf(qv[i + e] - m) * sc);
            *(v4h*)&Bq[tid][i] = h;                  // 8B store, 8B aligned (stride 144B)
        }
    }
    __syncthreads();

    const int lane = tid & 31, wave = tid >> 5;
    const int half = lane >> 4, l16 = lane & 15;

    for (int t = wave; t < 64; t += 8) {             // 4 j-tiles x 16 s-tiles
        const int tj = t & 3;
        const int ts = t >> 2;
        v8f acc = {};
        #pragma unroll
        for (int kk = 0; kk < DHEAD; kk += 32) {
            const _Float16* ap = &Ac[16 * tj + l16][kk + 8 * half];
            v8h alo = *(const v8h*)ap, ahi = *(const v8h*)(ap + 16);
            const _Float16* bp = &Bq[16 * ts + l16][kk + 8 * half];
            v8h blo = *(const v8h*)bp, bhi = *(const v8h*)(bp + 16);
            v16h A, B;
            #pragma unroll
            for (int e = 0; e < 8; ++e) {
                A[e] = alo[e]; A[e + 8] = ahi[e];
                B[e] = blo[e]; B[e + 8] = bhi[e];
            }
            acc = __builtin_amdgcn_wmma_f32_16x16x32_f16(false, A, false, B, (short)0, acc, false, false);
        }
        #pragma unroll
        for (int r = 0; r < 8; ++r) {
            int j = 16 * tj + 8 * half + r;
            out[base + (size_t)j * SEQ + s0 + 16 * ts + l16] = acc[r];
        }
    }
}

extern "C" void kernel_launch(void* const* d_in, const int* in_sizes, int n_in,
                              void* d_out, int out_size, void* d_ws, size_t ws_size,
                              hipStream_t stream) {
    (void)in_sizes; (void)n_in; (void)out_size; (void)ws_size;
    const float* q = (const float*)d_in[0];
    const float* k = (const float*)d_in[1];
    const float* v = (const float*)d_in[2];
    float* out = (float*)d_out;

    // workspace layout (floats): kmax[2048] | kinv[2048] | ctx[32*64*64]
    float* kmax = (float*)d_ws;
    float* kinv = kmax + NHEADS * DHEAD;
    float* ctx  = kinv + NHEADS * DHEAD;
    const int ctx_elems = NHEADS * DHEAD * DHEAD;    // 131072

    zero_ctx_kernel<<<(ctx_elems + 255) / 256, 256, 0, stream>>>(ctx, ctx_elems);
    kstats_kernel<<<NHEADS * DHEAD, 256, 0, stream>>>(k, kmax, kinv);
    context_kernel<<<dim3(NHEADS, SEQ / SCHUNK), 256, 0, stream>>>(k, v, kmax, kinv, ctx);
    output_kernel<<<dim3(NHEADS, SEQ / SC), 256, 0, stream>>>(q, ctx, out);
}
